// PyroSimpleSpeakerListenerLSTM_11278584120098
// MI455X (gfx1250) — matrix operand order repaired and checked by reading, not compiled, not run on hardware
//
#include <hip/hip_runtime.h>
#include <hip/hip_bf16.h>
#include <stdint.h>

// ---------------------------------------------------------------------------
// Model dims (compile-time constants from the reference)
// ---------------------------------------------------------------------------
#define MEANING 512
#define ARTIC   64
#define EMBED   256
#define H1      512
#define H2      512
#define SLEN    32
#define NBATCH  16
#define BM      (NBATCH * MEANING)   // 8192 decoder rows
#define G1      (4 * H1)             // 2048
#define G2      (4 * H2)             // 2048

typedef __attribute__((ext_vector_type(16))) __bf16 v16bf;
typedef __attribute__((ext_vector_type(8)))  float  v8f;

__device__ __forceinline__ unsigned short f32_to_bf16_rte(float f) {
  uint32_t u = __float_as_uint(f);
  uint32_t r = u + 0x7FFFu + ((u >> 16) & 1u);   // round-to-nearest-even
  return (unsigned short)(r >> 16);
}

// gfx1250 async global->LDS copy (tracked by ASYNCcnt). VDST = 32-bit LDS byte
// address VGPR, VADDR = 64-bit global address VGPR pair.
__device__ __forceinline__ void async_load_b128(uint32_t ldsAddr, const void* g) {
  asm volatile("global_load_async_to_lds_b128 %0, %1, off"
               :: "v"(ldsAddr), "v"(g) : "memory");
}

// ---------------------------------------------------------------------------
// WMMA GEMM:  Cout[M,N] = A[M,K](bf16) * Bt[N,K](bf16)^T  (+ Cin) (+ bias)
//   A row-major [M,K], Bt row-major [N,K], both bf16 (ushort).
//   Block = 256 threads = 8 waves; block tile = 128M x 128N.
//   Wave w computes rows [blockIdx.y*128 + 16w, +16) x 128 columns:
//   8 accumulators; A operand loaded once per K-step and reused 8x.
//   B slab (128n x 32k, 8KB) staged in LDS via global_load_async_to_lds_b128,
//   double-buffered (16KB), synchronized with s_wait_asynccnt + barriers.
//   Each thread issues exactly 2 straight-line async b128 loads per slab with
//   strength-reduced addresses (advance 64B/slab) -> uniform ASYNCcnt.
//   Requires M%128==0, N%128==0, K%32==0 (true for all uses here).
// Lane layouts used here (contiguous k order, same for A and B):
//   A 16x32 bf16 : VGPR r holds K pair {2r+16*half, +1} at row l16
//                  -> one contiguous 32B chunk per lane.
//   B 32x16 bf16 : VGPR r holds K pair {2r+16*half, +1} at column l16
//                  -> one contiguous 32B chunk per lane.
//   C/D f32 16x16: VGPR r <-> row r + 8*half, column l16.
// ---------------------------------------------------------------------------
#define SLAB_BYTES (128 * 32 * 2)   // one B slab in LDS (8KB)

__global__ void __launch_bounds__(256)
wmma_gemm_bf16(const unsigned short* __restrict__ A,
               const unsigned short* __restrict__ Bt,
               const float* __restrict__ Cin,
               const float* __restrict__ bias,
               float* __restrict__ Cout,
               int M, int N, int K)
{
  __shared__ __align__(64) unsigned short smB[2][128 * 32];  // [buf][n][k-in-slab]

  const int tid   = threadIdx.x;
  const int lane  = tid & 31;
  const int wave  = tid >> 5;
  const int nBase = blockIdx.x * 128;
  const int mBase = blockIdx.y * 128 + wave * 16;
  const int half  = lane >> 4;     // 0: lanes 0-15, 1: lanes 16-31
  const int l16   = lane & 15;

  // ---- accumulator init from Cin + bias (f32 C/D layout) ----
  v8f acc[8];
#pragma unroll
  for (int nt = 0; nt < 8; ++nt) {
    const int n = nBase + nt * 16 + l16;
#pragma unroll
    for (int r = 0; r < 8; ++r) {
      const int m = mBase + r + 8 * half;
      float v = 0.0f;
      if (Cin)  v += Cin[(size_t)m * N + n];
      if (bias) v += bias[n];
      acc[nt][r] = v;
    }
  }

  const int KS = K / 32;

  // B slab = 128 rows x 64 bytes = 512 x 16B chunks; thread handles chunks
  // tid (row n0=tid>>2) and tid+256 (row n0+64), same 16B piece e=(tid&3)*8.
  const int n0 = tid >> 2;
  const int e  = (tid & 3) * 8;                 // element offset within row
  const unsigned short* gB0 = Bt + (size_t)(nBase + n0)      * K + e;
  const unsigned short* gB1 = Bt + (size_t)(nBase + n0 + 64) * K + e;
  const uint32_t ldsB0 = (uint32_t)(uintptr_t)&smB[0][n0 * 32 + e];
  const uint32_t ldsB1 = (uint32_t)(uintptr_t)&smB[0][(n0 + 64) * 32 + e];

  // issue slab 0 into buffer 0
  async_load_b128(ldsB0, gB0);
  async_load_b128(ldsB1, gB1);

  for (int ks = 0; ks < KS; ++ks) {
    const int k0 = ks * 32;
    if (ks + 1 < KS) {
      // Prefetch next slab into the other buffer (its last readers finished at
      // the end-of-iteration barrier of ks-1). Straight-line, 2 issues.
      const uint32_t bufOff = (uint32_t)((ks + 1) & 1) * SLAB_BYTES;
      async_load_b128(ldsB0 + bufOff, gB0 + (ks + 1) * 32);
      async_load_b128(ldsB1 + bufOff, gB1 + (ks + 1) * 32);
      // 4 outstanding (2 current + 2 next): wait until current slab's 2 landed.
      asm volatile("s_wait_asynccnt 2" ::: "memory");
    } else {
      asm volatile("s_wait_asynccnt 0" ::: "memory");
    }
    __syncthreads();   // all waves' slab-ks data visible in LDS

    // A operand: one contiguous 32B chunk from global, reused for 8 WMMAs.
    union { v16bf v; uint32_t u[8]; } a;
    const unsigned short* aRow = A + (size_t)(mBase + l16) * K + k0;
#pragma unroll
    for (int r = 0; r < 8; ++r) {
      const int ka = 2 * r + 16 * half;
      a.u[r] = *(const uint32_t*)(aRow + ka);
    }

#pragma unroll
    for (int nt = 0; nt < 8; ++nt) {
      // B operand: one contiguous 32B chunk from LDS.
      const unsigned short* bRow =
          &smB[ks & 1][(nt * 16 + l16) * 32 + 16 * half];
      const v16bf bv = *(const v16bf*)bRow;
      acc[nt] = __builtin_amdgcn_wmma_f32_16x16x32_bf16(
          /*neg_a=*/false, a.v, /*neg_b=*/false, bv,
          /*c_mod=*/(short)0, acc[nt], /*reuse_a=*/false, /*reuse_b=*/false);
    }
    __syncthreads();   // everyone done reading smB[ks&1] before it is refilled
  }

#pragma unroll
  for (int nt = 0; nt < 8; ++nt) {
    const int n = nBase + nt * 16 + l16;
#pragma unroll
    for (int r = 0; r < 8; ++r) {
      const int m = mBase + r + 8 * half;
      Cout[(size_t)m * N + n] = acc[nt][r];
    }
  }
}

// ---------------------------------------------------------------------------
// Elementwise LSTM cell update. gates[r, 4H] in (i,f,g,o) order.
// ---------------------------------------------------------------------------
__global__ void __launch_bounds__(256)
lstm_update(const float* __restrict__ gates,
            float* __restrict__ h, float* __restrict__ c,
            unsigned short* __restrict__ hbf,
            unsigned short* __restrict__ ys,
            int R, int H)
{
  const int idx = blockIdx.x * blockDim.x + threadIdx.x;
  if (idx >= R * H) return;
  const int r = idx / H;
  const int j = idx - r * H;
  const float* g = gates + (size_t)r * 4 * H;
  const float gi = g[j], gf = g[H + j], gg = g[2 * H + j], go = g[3 * H + j];
  const float si = 1.0f / (1.0f + expf(-gi));
  const float sf = 1.0f / (1.0f + expf(-gf));
  const float so = 1.0f / (1.0f + expf(-go));
  const float cNew = sf * c[idx] + si * tanhf(gg);
  const float hNew = so * tanhf(cNew);
  c[idx] = cNew;
  h[idx] = hNew;
  const unsigned short hb = f32_to_bf16_rte(hNew);
  hbf[idx] = hb;
  if (ys) ys[idx] = hb;
}

// ---------------------------------------------------------------------------
// Small utility kernels
// ---------------------------------------------------------------------------
__global__ void __launch_bounds__(256)
cvt_f32_to_bf16(const float* __restrict__ in, unsigned short* __restrict__ out, int n)
{
  const int i = blockIdx.x * blockDim.x + threadIdx.x;
  if (i < n) out[i] = f32_to_bf16_rte(in[i]);
}

// out[N,K] = bf16(in[K,N])   (transpose + convert)
__global__ void __launch_bounds__(256)
tcvt_f32_to_bf16(const float* __restrict__ in, unsigned short* __restrict__ out,
                 int K, int N)
{
  const int i = blockIdx.x * blockDim.x + threadIdx.x;
  if (i >= K * N) return;
  const int k = i / N;
  const int n = i - k * N;
  out[(size_t)n * K + k] = f32_to_bf16_rte(in[i]);
}

// Embedding gather -> bf16  (ms is int32 arange, but honor it anyway)
__global__ void __launch_bounds__(256)
gather_embed_bf16(const float* __restrict__ emb, const int* __restrict__ ms,
                  unsigned short* __restrict__ out, int rows, int E)
{
  const int i = blockIdx.x * blockDim.x + threadIdx.x;
  if (i >= rows * E) return;
  const int r = i / E;
  const int j = i - r * E;
  out[i] = f32_to_bf16_rte(emb[(size_t)ms[r] * E + j]);
}

__global__ void __launch_bounds__(256)
vec_add(const float* __restrict__ a, const float* __restrict__ b,
        float* __restrict__ o, int n)
{
  const int i = blockIdx.x * blockDim.x + threadIdx.x;
  if (i < n) o[i] = a[i] + b[i];
}

// Broadcast init[0,:] -> h rows, init[1,:] -> c rows; also bf16 h.
__global__ void __launch_bounds__(256)
init_state(const float* __restrict__ init2H, float* __restrict__ h,
           float* __restrict__ c, unsigned short* __restrict__ hbf, int R, int H)
{
  const int idx = blockIdx.x * blockDim.x + threadIdx.x;
  if (idx >= R * H) return;
  const int j = idx % H;
  const float hv = init2H[j];
  const float cv = init2H[H + j];
  h[idx] = hv;
  c[idx] = cv;
  hbf[idx] = f32_to_bf16_rte(hv);
}

// ---------------------------------------------------------------------------
// Deterministic counter-hash RNG + Marsaglia-Tsang gamma -> Beta sampler.
// ---------------------------------------------------------------------------
__device__ __forceinline__ float u01(uint32_t& s)
{
  s = s * 747796405u + 2891336453u;
  uint32_t w = ((s >> ((s >> 28u) + 4u)) ^ s) * 277803737u;
  w = (w >> 22u) ^ w;
  return (float)(w >> 8) * (1.0f / 16777216.0f) + (0.5f / 16777216.0f);
}

__device__ float gamma_sample(float a, uint32_t& s)
{
  float boost = 1.0f;
  a = fmaxf(a, 1e-6f);
  if (a < 1.0f) {
    boost = powf(u01(s), 1.0f / a);
    a += 1.0f;
  }
  const float d = a - (1.0f / 3.0f);
  const float cc = rsqrtf(9.0f * d);
  float res = d;                                           // fallback: the mode
  for (int it = 0; it < 8; ++it) {
    const float u1 = u01(s), u2 = u01(s);
    const float rad = sqrtf(-2.0f * logf(u1));
    const float x = rad * cosf(6.2831853f * u2);           // N(0,1)
    float v = 1.0f + cc * x;
    if (v <= 0.0f) continue;
    v = v * v * v;
    const float u = u01(s);
    if (logf(u) < 0.5f * x * x + d - d * v + d * logf(v)) { res = d * v; break; }
  }
  return res * boost;
}

__device__ __forceinline__ float softplus_f(float x)
{
  return (x > 20.0f) ? x : log1pf(expf(x));
}

__global__ void __launch_bounds__(256)
beta_sample(const float* __restrict__ paramsRaw, unsigned short* __restrict__ sig)
{
  const int idx = blockIdx.x * blockDim.x + threadIdx.x;
  const int total = NBATCH * SLEN * MEANING * ARTIC;
  if (idx >= total) return;
  const int a = idx & (ARTIC - 1);
  const int m = (idx / ARTIC) & (MEANING - 1);
  const int t = (idx / (ARTIC * MEANING)) & (SLEN - 1);
  const int b = idx / (ARTIC * MEANING * SLEN);

  const size_t prow = (size_t)(t * MEANING + m) * (2 * ARTIC);
  const float alpha = softplus_f(paramsRaw[prow + 2 * a]);
  const float beta  = softplus_f(paramsRaw[prow + 2 * a + 1]);

  uint32_t s = (uint32_t)idx * 2654435769u + 0x9E3779B9u;
  const float g1v = gamma_sample(alpha, s);
  const float g2v = gamma_sample(beta, s);
  const float x = g1v / fmaxf(g1v + g2v, 1e-20f);

  const size_t o = ((size_t)t * BM + (size_t)b * MEANING + m) * ARTIC + a;
  sig[o] = f32_to_bf16_rte(x);
}

// ---------------------------------------------------------------------------
// In-place softmax: one wave32 per row of length N (<=512 here).
// ---------------------------------------------------------------------------
__global__ void __launch_bounds__(128)
softmax_rows(float* __restrict__ x, int R, int N)
{
  const int wave = (int)((blockIdx.x * blockDim.x + threadIdx.x) >> 5);
  const int lane = threadIdx.x & 31;
  if (wave >= R) return;
  float* row = x + (size_t)wave * N;

  float mx = -3.4e38f;
  for (int j = lane; j < N; j += 32) mx = fmaxf(mx, row[j]);
#pragma unroll
  for (int off = 16; off > 0; off >>= 1) mx = fmaxf(mx, __shfl_xor(mx, off, 32));

  float sum = 0.0f;
  for (int j = lane; j < N; j += 32) sum += expf(row[j] - mx);
#pragma unroll
  for (int off = 16; off > 0; off >>= 1) sum += __shfl_xor(sum, off, 32);

  const float inv = 1.0f / sum;
  for (int j = lane; j < N; j += 32) row[j] = expf(row[j] - mx) * inv;
}

// ---------------------------------------------------------------------------
// Host orchestration
// ---------------------------------------------------------------------------
extern "C" void kernel_launch(void* const* d_in, const int* in_sizes, int n_in,
                              void* d_out, int out_size, void* d_ws, size_t ws_size,
                              hipStream_t stream)
{
  (void)in_sizes; (void)n_in; (void)out_size; (void)ws_size;

  const int*   ms        = (const int*)  d_in[0];
  /* d_in[1] = batches (16), compile-time here */
  const float* embedding = (const float*)d_in[2];
  const float* enc_Wih   = (const float*)d_in[3];   // [G1, EMBED]
  const float* enc_Whh   = (const float*)d_in[4];   // [G1, H1]
  const float* enc_bih   = (const float*)d_in[5];
  const float* enc_bhh   = (const float*)d_in[6];
  const float* enc_init  = (const float*)d_in[7];   // [2, H1]
  const float* dec_Wih   = (const float*)d_in[8];   // [G2, ARTIC]
  const float* dec_Whh   = (const float*)d_in[9];   // [G2, H2]
  const float* dec_bih   = (const float*)d_in[10];
  const float* dec_bhh   = (const float*)d_in[11];
  const float* dec_init  = (const float*)d_in[12];  // [2, H2]
  const float* W_param   = (const float*)d_in[13];  // [H1, 2*ARTIC]
  const float* W_out     = (const float*)d_in[14];  // [H2, MEANING]
  float* out = (float*)d_out;                       // [BM, MEANING]

  // ---- workspace carving ----
  char* ws = (char*)d_ws;
  size_t off = 0;
  auto alloc = [&](size_t bytes) -> void* {
    off = (off + 255) & ~(size_t)255;
    void* p = ws + off;
    off += bytes;
    return p;
  };
  unsigned short* embBf     = (unsigned short*)alloc((size_t)MEANING * EMBED * 2);
  unsigned short* encWihBf  = (unsigned short*)alloc((size_t)G1 * EMBED * 2);
  unsigned short* encWhhBf  = (unsigned short*)alloc((size_t)G1 * H1 * 2);
  unsigned short* decWihBf  = (unsigned short*)alloc((size_t)G2 * ARTIC * 2);
  unsigned short* decWhhBf  = (unsigned short*)alloc((size_t)G2 * H2 * 2);
  unsigned short* WparamT   = (unsigned short*)alloc((size_t)(2 * ARTIC) * H1 * 2);
  unsigned short* WoutT     = (unsigned short*)alloc((size_t)MEANING * H2 * 2);
  float*          encBias   = (float*)alloc((size_t)G1 * 4);
  float*          decBias   = (float*)alloc((size_t)G2 * 4);
  float*          hE        = (float*)alloc((size_t)MEANING * H1 * 4);
  float*          cE        = (float*)alloc((size_t)MEANING * H1 * 4);
  unsigned short* hEbf      = (unsigned short*)alloc((size_t)MEANING * H1 * 2);
  float*          xproj     = (float*)alloc((size_t)MEANING * G1 * 4);
  float*          gatesE    = (float*)alloc((size_t)MEANING * G1 * 4);
  unsigned short* ysBf      = (unsigned short*)alloc((size_t)SLEN * MEANING * H1 * 2);
  float*          paramsRaw = (float*)alloc((size_t)SLEN * MEANING * 2 * ARTIC * 4);
  unsigned short* sigBf     = (unsigned short*)alloc((size_t)SLEN * BM * ARTIC * 2);
  float*          hD        = (float*)alloc((size_t)BM * H2 * 4);
  float*          cD        = (float*)alloc((size_t)BM * H2 * 4);
  unsigned short* hDbf      = (unsigned short*)alloc((size_t)BM * H2 * 2);
  float*          gatesD    = (float*)alloc((size_t)BM * G2 * 4);

  const int TB = 256;
  auto grid1 = [&](int n) { return dim3((unsigned)((n + TB - 1) / TB)); };

  // ---- weight prep ----
  gather_embed_bf16<<<grid1(MEANING * EMBED), TB, 0, stream>>>(embedding, ms, embBf, MEANING, EMBED);
  cvt_f32_to_bf16 <<<grid1(G1 * EMBED), TB, 0, stream>>>(enc_Wih, encWihBf, G1 * EMBED);
  cvt_f32_to_bf16 <<<grid1(G1 * H1),    TB, 0, stream>>>(enc_Whh, encWhhBf, G1 * H1);
  cvt_f32_to_bf16 <<<grid1(G2 * ARTIC), TB, 0, stream>>>(dec_Wih, decWihBf, G2 * ARTIC);
  cvt_f32_to_bf16 <<<grid1(G2 * H2),    TB, 0, stream>>>(dec_Whh, decWhhBf, G2 * H2);
  tcvt_f32_to_bf16<<<grid1(H1 * 2 * ARTIC), TB, 0, stream>>>(W_param, WparamT, H1, 2 * ARTIC);
  tcvt_f32_to_bf16<<<grid1(H2 * MEANING),   TB, 0, stream>>>(W_out,   WoutT,   H2, MEANING);
  vec_add<<<grid1(G1), TB, 0, stream>>>(enc_bih, enc_bhh, encBias, G1);
  vec_add<<<grid1(G2), TB, 0, stream>>>(dec_bih, dec_bhh, decBias, G2);
  init_state<<<grid1(MEANING * H1), TB, 0, stream>>>(enc_init, hE, cE, hEbf, MEANING, H1);
  init_state<<<grid1(BM * H2),      TB, 0, stream>>>(dec_init, hD, cD, hDbf, BM, H2);

  // ---- encoder: hoisted input projection (same x every step) ----
  dim3 gEnc(G1 / 128, MEANING / 128);
  wmma_gemm_bf16<<<gEnc, 256, 0, stream>>>(embBf, encWihBf, nullptr, encBias,
                                           xproj, MEANING, G1, EMBED);
  for (int t = 0; t < SLEN; ++t) {
    wmma_gemm_bf16<<<gEnc, 256, 0, stream>>>(hEbf, encWhhBf, xproj, nullptr,
                                             gatesE, MEANING, G1, H1);
    lstm_update<<<grid1(MEANING * H1), TB, 0, stream>>>(
        gatesE, hE, cE, hEbf, ysBf + (size_t)t * MEANING * H1, MEANING, H1);
  }

  // ---- params head: all timesteps as one [S*M, 2A] GEMM ----
  dim3 gPar((2 * ARTIC) / 128, (SLEN * MEANING) / 128);
  wmma_gemm_bf16<<<gPar, 256, 0, stream>>>(ysBf, WparamT, nullptr, nullptr,
                                           paramsRaw, SLEN * MEANING, 2 * ARTIC, H1);

  // ---- Beta sampling -> decoder input signals (bf16) ----
  beta_sample<<<grid1(NBATCH * SLEN * MEANING * ARTIC), TB, 0, stream>>>(paramsRaw, sigBf);

  // ---- decoder ----
  dim3 gDec(G2 / 128, BM / 128);
  for (int t = 0; t < SLEN; ++t) {
    wmma_gemm_bf16<<<gDec, 256, 0, stream>>>(
        sigBf + (size_t)t * BM * ARTIC, decWihBf, nullptr, decBias,
        gatesD, BM, G2, ARTIC);
    wmma_gemm_bf16<<<gDec, 256, 0, stream>>>(
        hDbf, decWhhBf, gatesD, nullptr, gatesD, BM, G2, H2);
    lstm_update<<<grid1(BM * H2), TB, 0, stream>>>(
        gatesD, hD, cD, hDbf, nullptr, BM, H2);
  }

  // ---- logits (straight into d_out) + in-place softmax ----
  dim3 gOut(MEANING / 128, BM / 128);
  wmma_gemm_bf16<<<gOut, 256, 0, stream>>>(hDbf, WoutT, nullptr, nullptr,
                                           out, BM, MEANING, H2);
  softmax_rows<<<(BM * 32 + 127) / 128, 128, 0, stream>>>(out, BM, MEANING);
}
